// GraphAutoEncoderModel_69363721831018
// MI455X (gfx1250) — matrix operand order repaired and checked by reading, not compiled, not run on hardware
//
#include <hip/hip_runtime.h>

// ---------------- types for WMMA ----------------
typedef __attribute__((ext_vector_type(16))) __bf16          v16bf;
typedef __attribute__((ext_vector_type(8)))  float           v8f;
typedef __attribute__((ext_vector_type(8)))  unsigned short  v8u16;
typedef __attribute__((ext_vector_type(16))) unsigned short  v16u16;

#define ALPHA_C 0.5f
#define BETA_C  0.04879016417f   // log(0.1/2 + 1)

static inline int blocks_for(long n, int t) { return (int)((n + t - 1) / t); }

__device__ __forceinline__ v16bf join_bf16(v8u16 lo, v8u16 hi) {
    v16u16 u = __builtin_shufflevector(lo, hi, 0, 1, 2, 3, 4, 5, 6, 7,
                                               8, 9, 10, 11, 12, 13, 14, 15);
    return __builtin_bit_cast(v16bf, u);
}

// ---------------- tiny utility kernels ----------------
__global__ void k_zero(float* __restrict__ p, long n) {
    long i = (long)blockIdx.x * blockDim.x + threadIdx.x;
    if (i < n) p[i] = 0.0f;
}

__global__ void k_fill(float* __restrict__ p, long n, float v) {
    long i = (long)blockIdx.x * blockDim.x + threadIdx.x;
    if (i < n) p[i] = v;
}

// ---------------- graph preprocessing ----------------
__global__ void k_build_edges(const long long* __restrict__ ei,
                              int* __restrict__ src, int* __restrict__ dst,
                              float* __restrict__ deg, int E, int E2) {
    int e = blockIdx.x * blockDim.x + threadIdx.x;
    if (e >= E2) return;
    int s, d;
    if (e < E) { s = (int)ei[e]; d = (int)ei[E + e]; }
    else       { s = d = e - E; }              // self loops
    src[e] = s; dst[e] = d;
    atomicAdd(&deg[d], 1.0f);
}

__global__ void k_dinv(float* __restrict__ deg, int n) {
    int i = blockIdx.x * blockDim.x + threadIdx.x;
    if (i >= n) return;
    float d = deg[i];
    deg[i] = (d > 0.0f) ? rsqrtf(d) : 0.0f;
}

__global__ void k_norm(const int* __restrict__ src, const int* __restrict__ dst,
                       const float* __restrict__ dinv, float* __restrict__ nrm, int E2) {
    int e = blockIdx.x * blockDim.x + threadIdx.x;
    if (e >= E2) return;
    nrm[e] = dinv[src[e]] * dinv[dst[e]];
}

// ---------------- f32 -> bf16 (RNE) conversion ----------------
__device__ __forceinline__ unsigned short f2bf(float f) {
    unsigned int u = __float_as_uint(f);
    u += 0x7FFFu + ((u >> 16) & 1u);
    return (unsigned short)(u >> 16);
}

// Activations: row-major [N, Kp], zero-padded K..Kp.
__global__ void k_cvtA(const float* __restrict__ A, unsigned short* __restrict__ Abf,
                       int N, int K, int Kp) {
    long i = (long)blockIdx.x * blockDim.x + threadIdx.x;
    if (i >= (long)N * Kp) return;
    int r = (int)(i / Kp), k = (int)(i % Kp);
    Abf[i] = (k < K) ? f2bf(A[(long)r * K + k]) : (unsigned short)0;
}

// Weights: packed directly into per-lane WMMA B-fragment order:
//   offset = ((mt*Kt + kt)*32 + lane)*16 + j,  element j of lane -> k = kt*32 + (lane/16)*16 + j,
//                                              m = mt*16 + (lane&15)
// so each lane's 16 bf16 values are contiguous (2 x b128 loads in the GEMM).
__global__ void k_cvtW_packed(const float* __restrict__ W, unsigned short* __restrict__ Wpk,
                              int K, int Kp, int M) {
    long i = (long)blockIdx.x * blockDim.x + threadIdx.x;
    if (i >= (long)Kp * M) return;
    int Kt = Kp >> 5;
    int j    = (int)(i & 15);
    int lane = (int)((i >> 4) & 31);
    long g   = i >> 9;                 // mt*Kt + kt
    int kt = (int)(g % Kt);
    int mt = (int)(g / Kt);
    int k = kt * 32 + ((lane >> 4) << 4) + j;
    int m = mt * 16 + (lane & 15);
    Wpk[i] = (k < K) ? f2bf(W[(long)k * M + m]) : (unsigned short)0;
}

// ---------------- WMMA GEMM: C[N,M] = A[N,Kp] * W[Kp,M] (bf16 -> f32 acc) ----------
// 4 waves / block; each wave owns a 32x16 output patch (2 row tiles x 1 col tile),
// reusing each B fragment for two v_wmma issues. The k-loop is branch-free: for the
// remainder row-block (N % 32 != 0) the second A pointer aliases the first and only
// its store is suppressed. All control flow is wave-uniform so EXEC is all-1s at
// every WMMA (ISA 7.12 requirement).
__global__ void k_wgemm(const unsigned short* __restrict__ A,
                        const unsigned short* __restrict__ Bpk,
                        float* __restrict__ C, int N, int Kp, int M) {
    int lane = threadIdx.x & 31;
    int wave = threadIdx.x >> 5;
    int Mt = M >> 4;
    int Kt = Kp >> 5;
    int mt = blockIdx.x * 4 + wave;        // 16-wide column tile
    if (mt >= Mt) return;                  // wave-uniform exit
    int rb = blockIdx.y;                   // 32-row block
    int half = lane >> 4;

    int row0 = rb * 32 + (lane & 15);
    int col  = mt * 16 + (lane & 15);
    bool has2 = (rb * 32 + 16) < N;        // wave-uniform (N % 16 == 0)

    const v8u16* ap0 = (const v8u16*)(A + (long)row0 * Kp + half * 16);
    const v8u16* ap1 = has2 ? (const v8u16*)(A + (long)(row0 + 16) * Kp + half * 16) : ap0;
    const v8u16* bp  = (const v8u16*)(Bpk + ((long)mt * Kt * 32 + lane) * 16);

    v8f c0 = {}, c1 = {};
    for (int kt = 0; kt < Kt; ++kt) {
        v16bf b  = join_bf16(bp[kt * 64], bp[kt * 64 + 1]);
        v16bf a0 = join_bf16(ap0[kt * 4], ap0[kt * 4 + 1]);
        v16bf a1 = join_bf16(ap1[kt * 4], ap1[kt * 4 + 1]);
        __builtin_prefetch((const void*)(bp + (kt + 2) * 64), 0, 3);   // global_prefetch_b8 (near scope)
        __builtin_prefetch((const void*)(ap0 + (kt + 2) * 4), 0, 3);
        c0 = __builtin_amdgcn_wmma_f32_16x16x32_bf16(
                 false, a0, false, b, (short)0, c0, false, false);
        c1 = __builtin_amdgcn_wmma_f32_16x16x32_bf16(
                 false, a1, false, b, (short)0, c1, false, false);
    }

    int rbase = rb * 32 + (half ? 8 : 0);
#pragma unroll
    for (int i = 0; i < 8; ++i)
        C[(long)(rbase + i) * M + col] = c0[i];
    if (has2) {
#pragma unroll
        for (int i = 0; i < 8; ++i)
            C[(long)(rbase + 16 + i) * M + col] = c1[i];
    }
}

// ---------------- normalized scatter-add propagation ----------------
// out[dst] += norm[e] * h[src], float4-vectorized; working set is L2-resident.
__global__ void k_prop(const float* __restrict__ h, const int* __restrict__ src,
                       const int* __restrict__ dst, const float* __restrict__ nrm,
                       float* __restrict__ out, int D4, long total) {
    long i = (long)blockIdx.x * blockDim.x + threadIdx.x;
    if (i >= total) return;
    int e = (int)(i / D4);
    int f = (int)(i % D4) * 4;
    int s = src[e], d = dst[e];
    float w = nrm[e];
    const float4 v = *(const float4*)(h + (long)s * (D4 * 4) + f);
    float* o = out + (long)d * (D4 * 4) + f;
    atomicAdd(o + 0, w * v.x);
    atomicAdd(o + 1, w * v.y);
    atomicAdd(o + 2, w * v.z);
    atomicAdd(o + 3, w * v.w);
}

// ---------------- elementwise ----------------
__global__ void k_bias_relu(const float* __restrict__ in, const float* __restrict__ bias,
                            float* __restrict__ out, long n, int D) {
    long i = (long)blockIdx.x * blockDim.x + threadIdx.x;
    if (i >= n) return;
    float v = in[i] + bias[i % D];
    out[i] = fmaxf(v, 0.0f);
}

__global__ void k_bias(const float* __restrict__ in, const float* __restrict__ bias,
                       float* __restrict__ out, long n, int D) {
    long i = (long)blockIdx.x * blockDim.x + threadIdx.x;
    if (i >= n) return;
    out[i] = in[i] + bias[i % D];
}

__global__ void k_combine1(float* __restrict__ h, const float* __restrict__ x0, long n) {
    long i = (long)blockIdx.x * blockDim.x + threadIdx.x;
    if (i >= n) return;
    h[i] = (1.0f - ALPHA_C) * h[i] + ALPHA_C * x0[i];
}

__global__ void k_combine2_relu(const float* __restrict__ h, const float* __restrict__ hw,
                                float* __restrict__ out, long n) {
    long i = (long)blockIdx.x * blockDim.x + threadIdx.x;
    if (i >= n) return;
    out[i] = fmaxf((1.0f - BETA_C) * h[i] + BETA_C * hw[i], 0.0f);
}

// ---------------- GATv2 ----------------
__device__ __forceinline__ void atomicMaxF(float* addr, float val) {
    unsigned int* ua = (unsigned int*)addr;
    unsigned int old = __float_as_uint(*addr);
    while (__uint_as_float(old) < val) {
        unsigned int prev = atomicCAS(ua, old, __float_as_uint(val));
        if (prev == old) break;
        old = prev;
    }
}

__global__ void k_gat_logits(const float* __restrict__ xl, const float* __restrict__ xr,
                             const float* __restrict__ att, const int* __restrict__ src,
                             const int* __restrict__ dst, float* __restrict__ alpha,
                             float* __restrict__ amax, int E2) {
    int i = blockIdx.x * blockDim.x + threadIdx.x;
    if (i >= E2 * 4) return;
    int e = i >> 2, h = i & 3;
    int s = src[e], d = dst[e];
    const float* pl = xl + (long)s * 128 + h * 32;
    const float* pr = xr + (long)d * 128 + h * 32;
    const float* pa = att + h * 32;
    float acc = 0.0f;
#pragma unroll
    for (int c = 0; c < 32; ++c) {
        float v = pl[c] + pr[c];
        v = (v > 0.0f) ? v : 0.2f * v;       // leaky_relu(0.2)
        acc += v * pa[c];
    }
    alpha[i] = acc;
    atomicMaxF(&amax[d * 4 + h], acc);
}

__global__ void k_gat_exp(float* __restrict__ alpha, const float* __restrict__ amax,
                          float* __restrict__ asum, const int* __restrict__ dst, int E2) {
    int i = blockIdx.x * blockDim.x + threadIdx.x;
    if (i >= E2 * 4) return;
    int e = i >> 2, h = i & 3;
    int d = dst[e];
    float a = expf(alpha[i] - amax[d * 4 + h]);
    alpha[i] = a;
    atomicAdd(&asum[d * 4 + h], a);
}

__global__ void k_gat_scatter(const float* __restrict__ xl, const float* __restrict__ alpha,
                              const float* __restrict__ asum, const int* __restrict__ src,
                              const int* __restrict__ dst, float* __restrict__ out, long total) {
    long i = (long)blockIdx.x * blockDim.x + threadIdx.x;
    if (i >= total) return;
    int e = (int)(i >> 7);
    int f = (int)(i & 127);
    int h = f >> 5;
    int s = src[e], d = dst[e];
    float w = alpha[e * 4 + h] / (asum[d * 4 + h] + 1e-16f);
    atomicAdd(&out[(long)d * 128 + f], w * xl[(long)s * 128 + f]);
}

// ---------------- host orchestration ----------------
extern "C" void kernel_launch(void* const* d_in, const int* in_sizes, int n_in,
                              void* d_out, int out_size, void* d_ws, size_t ws_size,
                              hipStream_t stream) {
    const float* x   = (const float*)d_in[0];
    const long long* ei = (const long long*)d_in[1];
    const float* W1  = (const float*)d_in[2];
    const float* b1  = (const float*)d_in[3];
    const float* W2  = (const float*)d_in[4];
    const float* b2  = (const float*)d_in[5];
    const float* W3  = (const float*)d_in[6];
    const float* b3  = (const float*)d_in[7];
    const float* W4  = (const float*)d_in[8];
    const float* b4  = (const float*)d_in[9];
    const float* Wl  = (const float*)d_in[10];
    const float* Wr  = (const float*)d_in[11];
    const float* att = (const float*)d_in[12];
    const float* bg  = (const float*)d_in[13];
    const float* Wc1 = (const float*)d_in[14];
    const float* W5  = (const float*)d_in[15];
    const float* b5  = (const float*)d_in[16];
    const float* Wc2 = (const float*)d_in[17];
    const float* Wo  = (const float*)d_in[18];
    const float* bo  = (const float*)d_in[19];
    float* out = (float*)d_out;

    const int N  = in_sizes[0] / 300;   // 50000
    const int E  = in_sizes[1] / 2;     // 800000
    const int E2 = E + N;               // self loops

    // ---- carve workspace ----
    size_t off = 0;
    auto alloc = [&](size_t bytes) {
        void* p = (char*)d_ws + off;
        off += (bytes + 255) & ~((size_t)255);
        return p;
    };
    int*   src2  = (int*)alloc((size_t)E2 * 4);
    int*   dst2  = (int*)alloc((size_t)E2 * 4);
    float* dinv  = (float*)alloc((size_t)N * 4);            // deg -> dinv in place
    float* nrm   = (float*)alloc((size_t)E2 * 4);
    float* B0    = (float*)alloc((size_t)N * 256 * 4);      // gemm out
    float* B1    = (float*)alloc((size_t)N * 256 * 4);      // x1 (persist)
    float* B2    = (float*)alloc((size_t)N * 128 * 4);      // x2 (persist)
    float* B3    = (float*)alloc((size_t)N * 256 * 4);      // current features
    float* B4    = (float*)alloc((size_t)N * 256 * 4);      // prop scratch / xr
    unsigned short* Abf = (unsigned short*)alloc((size_t)N * 320 * 2);
    unsigned short* Wpk = (unsigned short*)alloc((size_t)320 * 256 * 2);
    float* alpha = (float*)alloc((size_t)E2 * 4 * 4);
    float* amax  = (float*)alloc((size_t)N * 4 * 4);
    float* asum  = (float*)alloc((size_t)N * 4 * 4);

    const int T = 256;

    // ---- graph preprocessing ----
    k_zero<<<blocks_for(N, T), T, 0, stream>>>(dinv, N);
    k_build_edges<<<blocks_for(E2, T), T, 0, stream>>>(ei, src2, dst2, dinv, E, E2);
    k_dinv<<<blocks_for(N, T), T, 0, stream>>>(dinv, N);
    k_norm<<<blocks_for(E2, T), T, 0, stream>>>(src2, dst2, dinv, nrm, E2);

    auto gemm = [&](const float* A, int K, int Kp, int M, const float* W, float* C) {
        long na = (long)N * Kp;
        k_cvtA<<<blocks_for(na, T), T, 0, stream>>>(A, Abf, N, K, Kp);
        long nw = (long)Kp * M;
        k_cvtW_packed<<<blocks_for(nw, T), T, 0, stream>>>(W, Wpk, K, Kp, M);
        dim3 g(((M >> 4) + 3) / 4, (N + 31) / 32);
        k_wgemm<<<g, 128, 0, stream>>>(Abf, Wpk, C, N, Kp, M);
    };
    auto prop = [&](const float* h, int D, float* o) {
        long n = (long)N * D;
        k_zero<<<blocks_for(n, T), T, 0, stream>>>(o, n);
        long tot = (long)E2 * (D / 4);
        k_prop<<<blocks_for(tot, T), T, 0, stream>>>(h, src2, dst2, nrm, o, D / 4, tot);
    };

    // ---- layer 1: relu(prop(x@W1)+b1) -> B1 [N,256]
    gemm(x, 300, 320, 256, W1, B0);
    prop(B0, 256, B4);
    k_bias_relu<<<blocks_for((long)N * 256, T), T, 0, stream>>>(B4, b1, B1, (long)N * 256, 256);

    // ---- layer 2 -> B2 [N,128]
    gemm(B1, 256, 256, 128, W2, B0);
    prop(B0, 128, B4);
    k_bias_relu<<<blocks_for((long)N * 128, T), T, 0, stream>>>(B4, b2, B2, (long)N * 128, 128);

    // ---- layer 3 -> B3 [N,64]
    gemm(B2, 128, 128, 64, W3, B0);
    prop(B0, 64, B4);
    k_bias_relu<<<blocks_for((long)N * 64, T), T, 0, stream>>>(B4, b3, B3, (long)N * 64, 64);

    // ---- layer 4 -> B3 [N,32]
    gemm(B3, 64, 64, 32, W4, B0);
    prop(B0, 32, B4);
    k_bias_relu<<<blocks_for((long)N * 32, T), T, 0, stream>>>(B4, b4, B3, (long)N * 32, 32);

    // ---- GATv2: xl=B0, xr=B4 [N,128]
    gemm(B3, 32, 32, 128, Wl, B0);
    gemm(B3, 32, 32, 128, Wr, B4);
    k_fill<<<blocks_for((long)N * 4, T), T, 0, stream>>>(amax, (long)N * 4, -3.0e38f);
    k_zero<<<blocks_for((long)N * 4, T), T, 0, stream>>>(asum, (long)N * 4);
    k_gat_logits<<<blocks_for((long)E2 * 4, T), T, 0, stream>>>(B0, B4, att, src2, dst2, alpha, amax, E2);
    k_gat_exp<<<blocks_for((long)E2 * 4, T), T, 0, stream>>>(alpha, amax, asum, dst2, E2);
    k_zero<<<blocks_for((long)N * 128, T), T, 0, stream>>>(B3, (long)N * 128);
    k_gat_scatter<<<blocks_for((long)E2 * 128, T), T, 0, stream>>>(B0, alpha, asum, src2, dst2, B3, (long)E2 * 128);
    k_bias_relu<<<blocks_for((long)N * 128, T), T, 0, stream>>>(B3, bg, B3, (long)N * 128, 128);

    // ---- GCN2 #1 (x0 = x2 = B2, Wc1) -> B3 [N,128]
    prop(B3, 128, B4);
    k_combine1<<<blocks_for((long)N * 128, T), T, 0, stream>>>(B4, B2, (long)N * 128);
    gemm(B4, 128, 128, 128, Wc1, B0);
    k_combine2_relu<<<blocks_for((long)N * 128, T), T, 0, stream>>>(B4, B0, B3, (long)N * 128);

    // ---- layer: relu(prop(x@W5)+b5) -> B3 [N,256]
    gemm(B3, 128, 128, 256, W5, B0);
    prop(B0, 256, B4);
    k_bias_relu<<<blocks_for((long)N * 256, T), T, 0, stream>>>(B4, b5, B3, (long)N * 256, 256);

    // ---- GCN2 #2 (x0 = x1 = B1, Wc2) -> B3 [N,256]
    prop(B3, 256, B4);
    k_combine1<<<blocks_for((long)N * 256, T), T, 0, stream>>>(B4, B1, (long)N * 256);
    gemm(B4, 256, 256, 256, Wc2, B0);
    k_combine2_relu<<<blocks_for((long)N * 256, T), T, 0, stream>>>(B4, B0, B3, (long)N * 256);

    // ---- output: prop(x@Wo) + bo -> d_out [N,128]
    gemm(B3, 256, 256, 128, Wo, B0);
    prop(B0, 128, B4);
    k_bias<<<blocks_for((long)N * 128, T), T, 0, stream>>>(B4, bo, out, (long)N * 128, 128);
}
